// Mamba2Mixer_53635551593289
// MI455X (gfx1250) — matrix operand order repaired and checked by reading, not compiled, not run on hardware
//
#include <hip/hip_runtime.h>

// ---------------- problem constants (from reference) ----------------
constexpr int B_  = 2;
constexpr int L_  = 2048;
constexpr int H_  = 2048;
constexpr int I_  = 4096;
constexpr int N_  = 128;
constexpr int NH  = 64;
constexpr int P_  = 64;
constexpr int K_  = 4;
constexpr int CONV = I_ + 2 * N_;            // 4352
constexpr int PROJ = 2 * (I_ + N_) + NH;     // 8512
constexpr int ROWS = B_ * L_;                // 4096

// ---------------- types ----------------
typedef __attribute__((ext_vector_type(16))) __bf16       v16bf;
typedef __attribute__((ext_vector_type(8)))  float        v8f;
typedef __attribute__((ext_vector_type(4)))  unsigned int u32x4;
typedef __attribute__((ext_vector_type(8)))  int          i32x8;
typedef __attribute__((ext_vector_type(4)))  int          i32x4;

union Frag { v16bf v; u32x4 q[2]; };

__device__ __forceinline__ unsigned short f2bf(float f) {
    unsigned int u = __float_as_uint(f);
    u += 0x7FFFu + ((u >> 16) & 1u);         // round-to-nearest-even
    return (unsigned short)(u >> 16);
}
__device__ __forceinline__ float silu(float x) { return x / (1.f + __expf(-x)); }

// ---------------- TDM: async 2-D bf16 tile load, global -> LDS ----------------
// D# group0: [1:0]=count=1, [63:32]=lds_addr, [120:64]=global_addr, [127:126]=type(2)
// D# group1: [17:16]=data_size(1 => 2B), [79:48]=tensor_dim0, [111:80]=tensor_dim1,
//            [127:112]=tile_dim0, [143:128]=tile_dim1, [159:144]=tile_dim2(0),
//            [207:160]=tensor_dim0_stride
// groups 2/3: zero (2-D tile, tile_dim2/3 unused)
__device__ __forceinline__ void tdm_load_tile_bf16(unsigned ldsAddr,
                                                   const unsigned short* gsrc,
                                                   unsigned tileK, unsigned tileRows,
                                                   unsigned ldK) {
    unsigned long long ga = (unsigned long long)(size_t)gsrc;
    u32x4 g0;
    g0[0] = 1u;                                            // count=1, user desc
    g0[1] = ldsAddr;                                       // LDS byte address
    g0[2] = (unsigned)ga;                                  // global_addr[31:0]
    g0[3] = (unsigned)((ga >> 32) & 0x01FFFFFFull) | 0x80000000u; // [56:32] | type=2
    i32x8 g1;
    g1[0] = (int)(1u << 16);                               // data_size = 1 (2 bytes)
    g1[1] = (int)((ldK & 0xFFFFu) << 16);                  // tensor_dim0 lo
    g1[2] = (int)((ldK >> 16) | ((tileRows & 0xFFFFu) << 16)); // td0 hi | tensor_dim1 lo
    g1[3] = (int)((tileRows >> 16) | (tileK << 16));       // td1 hi | tile_dim0
    g1[4] = (int)(tileRows & 0xFFFFu);                     // tile_dim1 | tile_dim2=0
    g1[5] = (int)ldK;                                      // tensor_dim0_stride lo32
    g1[6] = 0;
    g1[7] = 0;
    i32x4 gz = (i32x4){0, 0, 0, 0};
#if defined(__clang_major__) && (__clang_major__ >= 23)
    i32x8 gz8 = (i32x8){0, 0, 0, 0, 0, 0, 0, 0};
    __builtin_amdgcn_tensor_load_to_lds(g0, g1, gz, gz, gz8, 0);
#else
    __builtin_amdgcn_tensor_load_to_lds(g0, g1, gz, gz, 0);
#endif
}

// ---------------- f32 -> bf16 convert ----------------
__global__ __launch_bounds__(256) void cvt_bf16(const float* __restrict__ in,
                                                unsigned short* __restrict__ out,
                                                long long n) {
    long long i = (long long)blockIdx.x * blockDim.x + threadIdx.x;
    if (i < n) out[i] = f2bf(in[i]);
}

// ---------------- TDM + LDS double-buffered WMMA GEMM: C[MxN] = A[MxK] * B[NxK]^T ----
// Block: 256 threads = 8 waves, 256(M) x 64(N) tile. Wave w: rows [w*32, w*32+32).
// Each wave: 2 A-frags x 4 B-frags = 8 WMMAs per K=32 step.
// A frag (16x32 bf16): lane m'=lane&15, h=lane>>4; e0..7 -> K=8h+e, e8..15 -> K=16+8h+(e-8)
// B frag (32x16 bf16): lane n'=lane&15, h=lane>>4; e0..7 -> K=8h+e, e8..15 -> K=16+8h+(e-8)
__global__ __launch_bounds__(256) void gemm_bf16_tdm(const unsigned short* __restrict__ A,
                                                     const unsigned short* __restrict__ B,
                                                     float* __restrict__ C,
                                                     int M, int N, int K) {
    __shared__ unsigned short shA[2][256 * 32];   // 2 x 16 KB
    __shared__ unsigned short shB[2][64 * 32];    // 2 x  4 KB

    const int lane = threadIdx.x & 31;
    const int wave = threadIdx.x >> 5;
    const int nStrips = N >> 6;
    const int mBase = (blockIdx.x / nStrips) << 8;   // 256 rows / block
    const int nBase = (blockIdx.x % nStrips) << 6;   // 64 cols / block

    const int mq = lane & 15;
    const int h  = lane >> 4;

    v8f acc[2][4];
    #pragma unroll
    for (int f = 0; f < 2; ++f)
        #pragma unroll
        for (int g = 0; g < 4; ++g)
            acc[f][g] = (v8f){0.f, 0.f, 0.f, 0.f, 0.f, 0.f, 0.f, 0.f};

    const int steps = K >> 5;

    if (threadIdx.x < 32) {   // one wave issues the DMAs (EXEC ignored by TDM)
        tdm_load_tile_bf16((unsigned)(size_t)&shA[0][0], A + (size_t)mBase * K, 32u, 256u, (unsigned)K);
        tdm_load_tile_bf16((unsigned)(size_t)&shB[0][0], B + (size_t)nBase * K, 32u, 64u,  (unsigned)K);
    }

    for (int s = 0; s < steps; ++s) {
        const int cur = s & 1;
        const int nxt = cur ^ 1;
        if (threadIdx.x < 32) {
            if (s + 1 < steps) {
                const int k0 = (s + 1) << 5;
                tdm_load_tile_bf16((unsigned)(size_t)&shA[nxt][0], A + (size_t)mBase * K + k0, 32u, 256u, (unsigned)K);
                tdm_load_tile_bf16((unsigned)(size_t)&shB[nxt][0], B + (size_t)nBase * K + k0, 32u, 64u,  (unsigned)K);
                __builtin_amdgcn_s_wait_tensorcnt(2);  // current pair done (in-order)
            } else {
                __builtin_amdgcn_s_wait_tensorcnt(0);
            }
        }
        __syncthreads();   // current tiles visible to all waves

        Frag a0, a1, fb[4];
        const unsigned short* ap = &shA[cur][(wave * 32 + mq) * 32];
        a0.q[0] = *(const u32x4*)(ap + 8 * h);
        a0.q[1] = *(const u32x4*)(ap + 16 + 8 * h);
        const unsigned short* ap1 = ap + 16 * 32;
        a1.q[0] = *(const u32x4*)(ap1 + 8 * h);
        a1.q[1] = *(const u32x4*)(ap1 + 16 + 8 * h);
        #pragma unroll
        for (int g = 0; g < 4; ++g) {
            const unsigned short* bp = &shB[cur][(g * 16 + mq) * 32 + 8 * h];
            fb[g].q[0] = *(const u32x4*)(bp);
            fb[g].q[1] = *(const u32x4*)(bp + 16);
        }

        #pragma unroll
        for (int g = 0; g < 4; ++g) {
            acc[0][g] = __builtin_amdgcn_wmma_f32_16x16x32_bf16(false, a0.v, false, fb[g].v, (short)0, acc[0][g], false, false);
            acc[1][g] = __builtin_amdgcn_wmma_f32_16x16x32_bf16(false, a1.v, false, fb[g].v, (short)0, acc[1][g], false, false);
        }
        __syncthreads();   // everyone done with buf[cur] before it is refilled
    }

    // D layout: VGPR r -> row (tileRow + r + 8h), col = n' (lane&15)
    #pragma unroll
    for (int f = 0; f < 2; ++f) {
        const int rowBase = mBase + wave * 32 + 16 * f + 8 * h;
        #pragma unroll
        for (int r = 0; r < 8; ++r) {
            size_t ro = (size_t)(rowBase + r) * N;
            #pragma unroll
            for (int g = 0; g < 4; ++g)
                C[ro + nBase + g * 16 + mq] = acc[f][g][r];
        }
    }
}

// ---------------- depthwise causal conv (K=4) + bias + SiLU + split ----------------
__global__ __launch_bounds__(256) void conv_silu(const float* __restrict__ proj,
                                                 const float* __restrict__ cw,
                                                 const float* __restrict__ cb,
                                                 float* __restrict__ xconv,
                                                 float* __restrict__ Bbuf,
                                                 float* __restrict__ Cbuf) {
    long long idx = (long long)blockIdx.x * blockDim.x + threadIdx.x;
    if (idx >= (long long)B_ * L_ * CONV) return;
    const int ch = (int)(idx % CONV);
    const int t  = (int)((idx / CONV) % L_);
    const int b  = (int)(idx / ((long long)CONV * L_));

    float acc = cb[ch];
    #pragma unroll
    for (int k = 0; k < K_; ++k) {
        int tt = t - (K_ - 1) + k;
        if (tt >= 0)
            acc = fmaf(cw[ch * K_ + k],
                       proj[((size_t)(b * L_ + tt)) * PROJ + I_ + ch], acc);
    }
    acc = silu(acc);
    const size_t row = (size_t)(b * L_ + t);
    if (ch < I_)            xconv[row * I_ + ch] = acc;
    else if (ch < I_ + N_)  Bbuf[row * N_ + (ch - I_)] = acc;
    else                    Cbuf[row * N_ + (ch - I_ - N_)] = acc;
}

// ---------------- dt = clip(softplus(dt+bias)), dA = exp(-exp(A_log)*dt) ----------------
__global__ __launch_bounds__(256) void dt_prep(const float* __restrict__ proj,
                                               const float* __restrict__ dt_bias,
                                               const float* __restrict__ A_log,
                                               float* __restrict__ dtbuf,
                                               float* __restrict__ dAbuf) {
    int idx = blockIdx.x * blockDim.x + threadIdx.x;
    if (idx >= ROWS * NH) return;
    const int hh = idx & (NH - 1);
    const int row = idx >> 6;
    float raw = proj[(size_t)row * PROJ + 2 * I_ + 2 * N_ + hh] + dt_bias[hh];
    float sp = (raw > 20.f) ? raw : log1pf(__expf(raw));
    sp = fminf(fmaxf(sp, 0.f), 1e10f);
    dtbuf[idx] = sp;
    dAbuf[idx] = __expf(-__expf(A_log[hh]) * sp);
}

// ---------------- SSD sequential scan (exact reformulation of chunked SSD) ------------
__global__ __launch_bounds__(256) void ssd_scan(const float* __restrict__ xconv,
                                                const float* __restrict__ Bbuf,
                                                const float* __restrict__ Cbuf,
                                                const float* __restrict__ dtbuf,
                                                const float* __restrict__ dAbuf,
                                                const float* __restrict__ Dv,
                                                float* __restrict__ ybuf) {
    const int b = blockIdx.x >> 6;
    const int h = blockIdx.x & 63;
    const int tid = threadIdx.x;
    const int p  = tid >> 2;
    const int ng = tid & 3;

    __shared__ float shB[N_];
    __shared__ float shC[N_];

    float s[32];
    #pragma unroll
    for (int j = 0; j < 32; ++j) s[j] = 0.f;
    const float Dh = Dv[h];

    for (int t = 0; t < L_; ++t) {
        const size_t row = (size_t)(b * L_ + t);
        if (tid < N_) shB[tid] = Bbuf[row * N_ + tid];
        else          shC[tid - N_] = Cbuf[row * N_ + (tid - N_)];
        __syncthreads();

        const float dtv = dtbuf[row * NH + h];
        const float dA  = dAbuf[row * NH + h];
        const float xv  = xconv[row * I_ + h * P_ + p];
        const float u   = dtv * xv;

        float partial = 0.f;
        const float* Bn = &shB[ng * 32];
        const float* Cn = &shC[ng * 32];
        #pragma unroll
        for (int j = 0; j < 32; ++j) {
            s[j] = fmaf(dA, s[j], u * Bn[j]);
            partial = fmaf(Cn[j], s[j], partial);
        }
        partial += __shfl_xor(partial, 1, 32);
        partial += __shfl_xor(partial, 2, 32);
        if (ng == 0) ybuf[row * I_ + h * P_ + p] = fmaf(Dh, xv, partial);
        __syncthreads();
    }
}

// ---------------- gated RMSNorm -> bf16 ----------------
__global__ __launch_bounds__(256) void gated_norm(const float* __restrict__ ybuf,
                                                  const float* __restrict__ proj,
                                                  const float* __restrict__ norm_w,
                                                  unsigned short* __restrict__ yfbf) {
    const int row = blockIdx.x;
    const int tid = threadIdx.x;
    __shared__ float rowv[I_];
    __shared__ float red[256];

    float local = 0.f;
    for (int i = tid; i < I_; i += 256) {
        float zv = proj[(size_t)row * PROJ + i];
        float yf = ybuf[(size_t)row * I_ + i] * silu(zv);
        rowv[i] = yf;
        local = fmaf(yf, yf, local);
    }
    red[tid] = local;
    __syncthreads();
    for (int s = 128; s > 0; s >>= 1) {
        if (tid < s) red[tid] += red[tid + s];
        __syncthreads();
    }
    const float scale = rsqrtf(red[0] / (float)I_ + 1e-5f);
    for (int i = tid; i < I_; i += 256)
        yfbf[(size_t)row * I_ + i] = f2bf(rowv[i] * scale * norm_w[i]);
}

// ---------------- launcher ----------------
extern "C" void kernel_launch(void* const* d_in, const int* in_sizes, int n_in,
                              void* d_out, int out_size, void* d_ws, size_t ws_size,
                              hipStream_t stream) {
    (void)in_sizes; (void)n_in; (void)out_size; (void)ws_size;
    const float* hs      = (const float*)d_in[0];
    const float* W_in    = (const float*)d_in[1];
    const float* conv_w  = (const float*)d_in[2];
    const float* conv_b  = (const float*)d_in[3];
    const float* dt_bias = (const float*)d_in[4];
    const float* A_log   = (const float*)d_in[5];
    const float* Dvec    = (const float*)d_in[6];
    const float* norm_w  = (const float*)d_in[7];
    const float* W_out   = (const float*)d_in[8];
    float* out = (float*)d_out;

    char* ws = (char*)d_ws;
    size_t off = 0;
    auto take = [&](size_t bytes) { char* p = ws + off; off += (bytes + 255) & ~(size_t)255; return p; };
    unsigned short* hs_bf   = (unsigned short*)take((size_t)ROWS * H_ * 2);
    unsigned short* Win_bf  = (unsigned short*)take((size_t)PROJ * H_ * 2);
    unsigned short* Wout_bf = (unsigned short*)take((size_t)H_ * I_ * 2);
    float* proj  = (float*)take((size_t)ROWS * PROJ * 4);
    float* xconv = (float*)take((size_t)ROWS * I_ * 4);
    float* Bbuf  = (float*)take((size_t)ROWS * N_ * 4);
    float* Cbuf  = (float*)take((size_t)ROWS * N_ * 4);
    float* dtb   = (float*)take((size_t)ROWS * NH * 4);
    float* dAb   = (float*)take((size_t)ROWS * NH * 4);
    float* ybuf  = (float*)take((size_t)ROWS * I_ * 4);
    unsigned short* yfbf = (unsigned short*)take((size_t)ROWS * I_ * 2);

    // 1) bf16 conversions
    {
        long long n = (long long)ROWS * H_;
        cvt_bf16<<<(unsigned)((n + 255) / 256), 256, 0, stream>>>(hs, hs_bf, n);
        n = (long long)PROJ * H_;
        cvt_bf16<<<(unsigned)((n + 255) / 256), 256, 0, stream>>>(W_in, Win_bf, n);
        n = (long long)H_ * I_;
        cvt_bf16<<<(unsigned)((n + 255) / 256), 256, 0, stream>>>(W_out, Wout_bf, n);
    }

    // 2) proj = hs @ W_in^T   (M=4096, N=8512, K=2048): 16 x 133 blocks of 256x64
    gemm_bf16_tdm<<<(ROWS / 256) * (PROJ / 64), 256, 0, stream>>>(hs_bf, Win_bf, proj, ROWS, PROJ, H_);

    // 3) depthwise conv + SiLU + split
    {
        long long n = (long long)B_ * L_ * CONV;
        conv_silu<<<(unsigned)((n + 255) / 256), 256, 0, stream>>>(proj, conv_w, conv_b, xconv, Bbuf, Cbuf);
    }

    // 4) dt / dA prep
    dt_prep<<<(ROWS * NH + 255) / 256, 256, 0, stream>>>(proj, dt_bias, A_log, dtb, dAb);

    // 5) SSD scan: one block per (b, h)
    ssd_scan<<<B_ * NH, 256, 0, stream>>>(xconv, Bbuf, Cbuf, dtb, dAb, Dvec, ybuf);

    // 6) gated RMSNorm -> bf16
    gated_norm<<<ROWS, 256, 0, stream>>>(ybuf, proj, norm_w, yfbf);

    // 7) out = yf @ W_out^T   (M=4096, N=2048, K=4096): 16 x 32 blocks
    gemm_bf16_tdm<<<(ROWS / 256) * (H_ / 64), 256, 0, stream>>>(yfbf, Wout_bf, out, ROWS, H_, I_);
}
